// TrorYongASRModel_75952201663175
// MI455X (gfx1250) — matrix operand order repaired and checked by reading, not compiled, not run on hardware
//
#include <hip/hip_runtime.h>

// ---------------------------------------------------------------------------
// Types for CDNA5 WMMA (wave32, v_wmma_f32_16x16x32_bf16)
// ---------------------------------------------------------------------------
typedef __bf16 bf16;
typedef __bf16 bf16x16 __attribute__((ext_vector_type(16)));
typedef __bf16 bf16x8  __attribute__((ext_vector_type(8)));
typedef float  f32x8   __attribute__((ext_vector_type(8)));

union FragU { bf16x16 v; bf16x8 h[2]; };

// Model dims
#define BSZ   4
#define TIN   1500
#define TENC  750
#define NE    512
#define FF    2048
#define EH    8
#define EHD   64
#define DH    16
#define DHD   32
#define LDEC  448
#define NVOC  51904
#define M_ENC (BSZ * TENC)   // 3000
#define M_DEC (BSZ * LDEC)   // 1792

// LDS tile row stride: 32 bf16 payload padded to 40 elems (80B) ->
// conflict-free 16-row fragment reads, 16B-aligned for b128.
#define LROW 40

__device__ __forceinline__ float gelu_f(float x) {
  return 0.5f * x * (1.0f + erff(x * 0.70710678118654752440f));
}

// ---------------------------------------------------------------------------
// fp32 -> bf16 conversion (weights / staging)
// ---------------------------------------------------------------------------
__global__ void f32_to_bf16_kernel(const float* __restrict__ in,
                                   bf16* __restrict__ out, long long n) {
  long long i = (long long)blockIdx.x * blockDim.x + threadIdx.x;
  long long stride = (long long)gridDim.x * blockDim.x;
  for (; i < n; i += stride) out[i] = (bf16)in[i];
}

// ---------------------------------------------------------------------------
// WMMA GEMM:  C[M,N] = A[M,K](bf16) * W[N,K](bf16)^T  (+bias, +resid, gelu)
// Block: 256 threads = 8 waves as 2(M) x 4(N); wave tile 32x32; block 64x128.
// A/B tiles staged to LDS with async DMA (global_load_async_to_lds_b128),
// double-buffered; s_wait_asynccnt + barrier per K-step.
// flags bit0 = gelu.  Cf (fp32) and/or Cb (bf16) outputs, either may be null.
// ---------------------------------------------------------------------------
__device__ __forceinline__ void epi_store(float v, int r, int c, int M, int N,
                                          const float* bias, const float* resid,
                                          float* Cf, bf16* Cb, int flags) {
  if (r >= M || c >= N) return;
  size_t idx = (size_t)r * N + c;
  if (bias)  v += bias[c];
  if (resid) v += resid[idx];
  if (flags & 1) v = gelu_f(v);
  if (Cf) Cf[idx] = v;
  if (Cb) Cb[idx] = (bf16)v;
}

__global__ __launch_bounds__(256) void gemm_bf16_wmma_kernel(
    const bf16* __restrict__ A, const bf16* __restrict__ W,
    const float* __restrict__ bias, const float* __restrict__ resid,
    float* __restrict__ Cf, bf16* __restrict__ Cb,
    int M, int N, int K, int flags) {
  __shared__ bf16 sA[2][64 * LROW];    // 64 rows  x 32 K (padded)
  __shared__ bf16 sB[2][128 * LROW];   // 128 cols x 32 K (padded)

  const int t = threadIdx.x;
  const int lane = t & 31;
  const int wave = t >> 5;
  const int wm = wave >> 2;              // 0..1
  const int wn = wave & 3;               // 0..3
  const int row0 = blockIdx.y * 64;
  const int col0 = blockIdx.x * 128;
  const int lhi = lane >> 4;             // 0/1
  const int llo = lane & 15;

  // ---- cooperative async loader: thread t owns (row t>>2, 16B segment t&3) ----
  const int lr = t >> 2;                 // 0..63
  const int ls = t & 3;                  // 0..3 (8 bf16 each)
  int arow = row0 + lr;          arow = arow < M ? arow : M - 1;
  int bcol0 = col0 + lr;         bcol0 = bcol0 < N ? bcol0 : N - 1;
  int bcol1 = col0 + 64 + lr;    bcol1 = bcol1 < N ? bcol1 : N - 1;
  const unsigned aoff  = ((unsigned)arow  * (unsigned)K + (unsigned)ls * 8u) * 2u;
  const unsigned boff0 = ((unsigned)bcol0 * (unsigned)K + (unsigned)ls * 8u) * 2u;
  const unsigned boff1 = ((unsigned)bcol1 * (unsigned)K + (unsigned)ls * 8u) * 2u;

  unsigned lA[2], lB0[2], lB1[2];
#pragma unroll
  for (int bi = 0; bi < 2; ++bi) {
    lA[bi]  = (unsigned)(size_t)&sA[bi][lr * LROW + ls * 8];
    lB0[bi] = (unsigned)(size_t)&sB[bi][lr * LROW + ls * 8];
    lB1[bi] = (unsigned)(size_t)&sB[bi][(64 + lr) * LROW + ls * 8];
  }

  auto issue = [&](int bi, int kk) {
    unsigned kb = (unsigned)(kk * 2);
    asm volatile("global_load_async_to_lds_b128 %0, %1, %2"
                 :: "v"(lA[bi]),  "v"(aoff  + kb), "s"(A) : "memory");
    asm volatile("global_load_async_to_lds_b128 %0, %1, %2"
                 :: "v"(lB0[bi]), "v"(boff0 + kb), "s"(W) : "memory");
    asm volatile("global_load_async_to_lds_b128 %0, %1, %2"
                 :: "v"(lB1[bi]), "v"(boff1 + kb), "s"(W) : "memory");
    __builtin_prefetch((const char*)A + aoff  + kb + 128, 0, 0);
    __builtin_prefetch((const char*)W + boff0 + kb + 128, 0, 0);
  };

  f32x8 acc00 = {0,0,0,0,0,0,0,0};
  f32x8 acc01 = {0,0,0,0,0,0,0,0};
  f32x8 acc10 = {0,0,0,0,0,0,0,0};
  f32x8 acc11 = {0,0,0,0,0,0,0,0};

  const int nk = K >> 5;
  issue(0, 0);
  for (int i = 0; i < nk; ++i) {
    asm volatile("s_wait_asynccnt 0" ::: "memory");  // own tile writes landed
    __syncthreads();                                  // everyone's landed; prev reads done
    if (i + 1 < nk) issue((i + 1) & 1, (i + 1) << 5); // overlap DMA with WMMA below
    const int buf = i & 1;
    const bf16* pa0 = &sA[buf][(wm * 32 + llo) * LROW + lhi * 8];
    const bf16* pa1 = pa0 + 16 * LROW;
    const bf16* pb0 = &sB[buf][(wn * 32 + llo) * LROW + lhi * 8];
    const bf16* pb1 = pb0 + 16 * LROW;
    FragU a0, a1, b0, b1;
    // A frag: halves 0..7 = K[lhi*8..+7], halves 8..15 = K[lhi*8+16..+23]
    a0.h[0] = *(const bf16x8*)(pa0);  a0.h[1] = *(const bf16x8*)(pa0 + 16);
    a1.h[0] = *(const bf16x8*)(pa1);  a1.h[1] = *(const bf16x8*)(pa1 + 16);
    // B frag: halves 0..7 = K[lhi*8..+7], halves 8..15 = K[lhi*8+16..+23]
    b0.h[0] = *(const bf16x8*)(pb0);  b0.h[1] = *(const bf16x8*)(pb0 + 16);
    b1.h[0] = *(const bf16x8*)(pb1);  b1.h[1] = *(const bf16x8*)(pb1 + 16);
    acc00 = __builtin_amdgcn_wmma_f32_16x16x32_bf16(false, a0.v, false, b0.v, (short)0, acc00, false, false);
    acc01 = __builtin_amdgcn_wmma_f32_16x16x32_bf16(false, a0.v, false, b1.v, (short)0, acc01, false, false);
    acc10 = __builtin_amdgcn_wmma_f32_16x16x32_bf16(false, a1.v, false, b0.v, (short)0, acc10, false, false);
    acc11 = __builtin_amdgcn_wmma_f32_16x16x32_bf16(false, a1.v, false, b1.v, (short)0, acc11, false, false);
  }

  // D layout: VGPR e -> row (lhi*8 + e), col llo.
  const int wr0 = row0 + wm * 32;
  const int wc0 = col0 + wn * 32;
#pragma unroll
  for (int e = 0; e < 8; ++e) {
    int rr = wr0 + lhi * 8 + e;
    int cc = wc0 + llo;
    epi_store(acc00[e], rr,      cc,      M, N, bias, resid, Cf, Cb, flags);
    epi_store(acc01[e], rr,      cc + 16, M, N, bias, resid, Cf, Cb, flags);
    epi_store(acc10[e], rr + 16, cc,      M, N, bias, resid, Cf, Cb, flags);
    epi_store(acc11[e], rr + 16, cc + 16, M, N, bias, resid, Cf, Cb, flags);
  }
}

// ---------------------------------------------------------------------------
// rms over rows of length 512 -> bf16
// ---------------------------------------------------------------------------
__global__ __launch_bounds__(256) void rms512_kernel(const float* __restrict__ In,
                                                     bf16* __restrict__ Out, int R) {
  __shared__ float red[256];
  int r = blockIdx.x, t = threadIdx.x;
  if (r >= R) return;
  const float* row = In + (size_t)r * NE;
  float a = row[t], b = row[t + 256];
  red[t] = a * a + b * b;
  __syncthreads();
  for (int o = 128; o > 0; o >>= 1) { if (t < o) red[t] += red[t + o]; __syncthreads(); }
  float sc = rsqrtf(red[0] / (float)NE + 1e-6f);
  bf16* orow = Out + (size_t)r * NE;
  orow[t] = (bf16)(a * sc);
  orow[t + 256] = (bf16)(b * sc);
}

// ---------------------------------------------------------------------------
// split heads (b,l,H*D) -> (b,H,l,D), optional rms over D.  block=64, grid (L,H,B)
// ---------------------------------------------------------------------------
__global__ __launch_bounds__(64) void split_heads_kernel(const float* __restrict__ In,
                                                         float* __restrict__ Out,
                                                         int L, int H, int D, int do_rms) {
  __shared__ float red[64];
  int l = blockIdx.x, h = blockIdx.y, b = blockIdx.z, t = threadIdx.x;
  float v = 0.f;
  if (t < D) v = In[((size_t)b * L + l) * (H * D) + (size_t)h * D + t];
  if (do_rms) {
    red[t] = (t < D) ? v * v : 0.f;
    __syncthreads();
    for (int o = 32; o > 0; o >>= 1) { if (t < o) red[t] += red[t + o]; __syncthreads(); }
    v *= rsqrtf(red[0] / (float)D + 1e-6f);
  }
  if (t < D) Out[(((size_t)b * H + h) * L + l) * D + t] = v;
}

// ---------------------------------------------------------------------------
// rope + split + rms for decoder (H=16, D=32). block=32 (one wave), grid (L,16,B)
// kmode=1 -> key positions (shifted by one).
// ---------------------------------------------------------------------------
__global__ __launch_bounds__(32) void rope_split_rms_kernel(const float* __restrict__ In,
                                                            float* __restrict__ Out,
                                                            int L, int kmode) {
  int l = blockIdx.x, h = blockIdx.y, b = blockIdx.z, t = threadIdx.x;
  const float* row = In + ((size_t)b * L + l) * NE + (size_t)h * DHD;
  int p = t & 15;
  float inv = powf(10000.0f, -(float)p / 16.0f);
  int pos = kmode ? (l > 0 ? l - 1 : 0) : l;
  float ang = (float)pos * inv;
  float c = cosf(ang), sn = sinf(ang);
  float x1 = row[p], x2 = row[p + 16];
  float val = (t < 16) ? (x1 * c + x2 * sn) : (x2 * c - x1 * sn);
  float ss = val * val;
  for (int m = 16; m > 0; m >>= 1) ss += __shfl_xor(ss, m, 32);
  val *= rsqrtf(ss / (float)DHD + 1e-6f);
  Out[(((size_t)b * DH + h) * L + l) * DHD + t] = val;
}

// ---------------------------------------------------------------------------
// Fused attention (softmax + AV + optional decorrelation), fp32 in, bf16 out.
// grid (Lq, H, B), block 256.  out layout (b, lq, H*D) for the O-projection GEMM.
// ---------------------------------------------------------------------------
__global__ __launch_bounds__(256) void attn_kernel(
    const float* __restrict__ Q, const float* __restrict__ Km,
    const float* __restrict__ V, const int* __restrict__ ids,
    bf16* __restrict__ Out, int H, int Lq, int Lk, int D,
    int causal, int decor, float scale) {
  __shared__ float sm[768];
  __shared__ float qb[64];
  __shared__ float vq[64];
  __shared__ float red[256];
  int b = blockIdx.z, h = blockIdx.y, qi = blockIdx.x, t = threadIdx.x;
  size_t qoff  = (((size_t)b * H + h) * Lq + qi) * D;
  size_t kbase = ((size_t)b * H + h) * (size_t)Lk * D;
  if (t < D) qb[t] = Q[qoff + t];
  if (decor && t < D) vq[t] = V[kbase + (size_t)qi * D + t];
  __syncthreads();

  float lmax = -1e30f;
  for (int kk = t; kk < Lk; kk += 256) {
    const float* kr = Km + kbase + (size_t)kk * D;
    float d0 = 0.f;
    for (int dd = 0; dd < D; ++dd) d0 += qb[dd] * kr[dd];
    d0 *= scale;
    if (causal && kk > qi) d0 = -1e30f;
    if (ids && ids[(size_t)b * Lk + kk] == 0) d0 = -1e30f;
    sm[kk] = d0;
    lmax = fmaxf(lmax, d0);
  }
  red[t] = lmax;
  __syncthreads();
  for (int o = 128; o > 0; o >>= 1) { if (t < o) red[t] = fmaxf(red[t], red[t + o]); __syncthreads(); }
  float m = red[0];
  __syncthreads();

  float lsum = 0.f;
  for (int kk = t; kk < Lk; kk += 256) { float e = __expf(sm[kk] - m); sm[kk] = e; lsum += e; }
  red[t] = lsum;
  __syncthreads();
  for (int o = 128; o > 0; o >>= 1) { if (t < o) red[t] += red[t + o]; __syncthreads(); }
  float inv = 1.0f / red[0];
  __syncthreads();

  float od = 0.f;
  if (t < D) {
    for (int kk = 0; kk < Lk; ++kk) od += sm[kk] * V[kbase + (size_t)kk * D + t];
    od *= inv;
  }

  if (decor) {
    red[t] = (t < D) ? vq[t] * vq[t] : 0.f;
    __syncthreads();
    for (int o = 128; o > 0; o >>= 1) { if (t < o) red[t] += red[t + o]; __syncthreads(); }
    float nrm = fmaxf(sqrtf(red[0]), 1e-9f);
    __syncthreads();
    red[t] = (t < D) ? od * (vq[t] / nrm) : 0.f;
    __syncthreads();
    for (int o = 128; o > 0; o >>= 1) { if (t < o) red[t] += red[t + o]; __syncthreads(); }
    float dt = red[0];
    __syncthreads();
    if (t < D) od -= dt * (vq[t] / nrm);
  }
  if (t < D) Out[((size_t)b * Lq + qi) * (H * D) + (size_t)h * D + t] = (bf16)od;
}

// ---------------------------------------------------------------------------
// conv1: (B,80,1500) -> gelu -> (B,512,1500), k=3, pad 1, stride 1
// ---------------------------------------------------------------------------
__global__ __launch_bounds__(256) void conv1_kernel(const float* __restrict__ X,
                                                    const float* __restrict__ Wt,
                                                    const float* __restrict__ Bi,
                                                    float* __restrict__ Y) {
  size_t i = (size_t)blockIdx.x * blockDim.x + threadIdx.x;
  if (i >= (size_t)BSZ * NE * TIN) return;
  int t  = (int)(i % TIN);
  int co = (int)((i / TIN) % NE);
  int b  = (int)(i / ((size_t)TIN * NE));
  const float* wr = Wt + (size_t)co * 80 * 3;
  const float* mr = X + (size_t)b * 80 * TIN;
  float acc = Bi[co];
  for (int ci = 0; ci < 80; ++ci) {
    const float* m = mr + (size_t)ci * TIN;
    const float* w = wr + ci * 3;
    if (t > 0)       acc += m[t - 1] * w[0];
    acc += m[t] * w[1];
    if (t < TIN - 1) acc += m[t + 1] * w[2];
  }
  Y[i] = gelu_f(acc);
}

// conv2: (B,512,1500) -> gelu -> (B,512,750), k=3, pad 1, stride 2
__global__ __launch_bounds__(256) void conv2_kernel(const float* __restrict__ X,
                                                    const float* __restrict__ Wt,
                                                    const float* __restrict__ Bi,
                                                    float* __restrict__ Y) {
  size_t i = (size_t)blockIdx.x * blockDim.x + threadIdx.x;
  if (i >= (size_t)BSZ * NE * TENC) return;
  int t  = (int)(i % TENC);
  int co = (int)((i / TENC) % NE);
  int b  = (int)(i / ((size_t)TENC * NE));
  const float* wr = Wt + (size_t)co * NE * 3;
  const float* mr = X + (size_t)b * NE * TIN;
  int ti = 2 * t;
  float acc = Bi[co];
  for (int ci = 0; ci < NE; ++ci) {
    const float* m = mr + (size_t)ci * TIN;
    const float* w = wr + ci * 3;
    if (ti > 0) acc += m[ti - 1] * w[0];
    acc += m[ti] * w[1];
    acc += m[ti + 1] * w[2];
  }
  Y[i] = gelu_f(acc);
}

// transpose (B,512,750)->(B,750,512) and add sinusoidal positions
__global__ __launch_bounds__(256) void pos_kernel(const float* __restrict__ X,
                                                  float* __restrict__ Y) {
  size_t i = (size_t)blockIdx.x * blockDim.x + threadIdx.x;
  if (i >= (size_t)BSZ * TENC * NE) return;
  int c = (int)(i % NE);
  int t = (int)((i / NE) % TENC);
  int b = (int)(i / ((size_t)NE * TENC));
  const float inc = 9.210340371976184f / 255.0f;   // ln(10000)/(256-1)
  int j = c & 255;
  float ang = (float)t * expf(-inc * (float)j);
  float pos = (c < 256) ? sinf(ang) : cosf(ang);
  Y[i] = X[((size_t)b * NE + c) * TENC + t] + pos;
}

// ctx = embed[ids]*sqrt(512) (+pos_embed shifted); q0 = pos_embed broadcast
__global__ __launch_bounds__(256) void ctx_kernel(const int* __restrict__ ids,
                                                  const float* __restrict__ emb,
                                                  const float* __restrict__ pe,
                                                  float* __restrict__ ctx,
                                                  float* __restrict__ q0) {
  size_t i = (size_t)blockIdx.x * blockDim.x + threadIdx.x;
  if (i >= (size_t)BSZ * LDEC * NE) return;
  int c = (int)(i % NE);
  int l = (int)((i / NE) % LDEC);
  int b = (int)(i / ((size_t)NE * LDEC));
  int id = ids[(size_t)b * LDEC + l];
  float v = emb[(size_t)id * NE + c] * 22.62741699796952f;   // sqrt(512)
  if (l > 0) v += pe[(size_t)(l - 1) * NE + c];
  ctx[i] = v;
  q0[i] = pe[(size_t)l * NE + c];
}

// ---------------------------------------------------------------------------
// Host orchestration
// ---------------------------------------------------------------------------
static inline void* arena_alloc(char*& p, size_t bytes) {
  void* r = (void*)p;
  p += (bytes + 255) & ~(size_t)255;
  return r;
}

static void gemm(hipStream_t s, const bf16* A, const bf16* W, const float* bias,
                 const float* resid, float* Cf, bf16* Cb, int M, int N, int K,
                 int flags) {
  dim3 g((N + 127) / 128, (M + 63) / 64);
  gemm_bf16_wmma_kernel<<<g, 256, 0, s>>>(A, W, bias, resid, Cf, Cb, M, N, K, flags);
}

extern "C" void kernel_launch(void* const* d_in, const int* in_sizes, int n_in,
                              void* d_out, int out_size, void* d_ws, size_t ws_size,
                              hipStream_t stream) {
  (void)in_sizes; (void)n_in; (void)out_size; (void)ws_size;
  const float* mels    = (const float*)d_in[0];
  const int*   ids     = (const int*)d_in[1];
  const float* conv1_w = (const float*)d_in[2];
  const float* conv1_b = (const float*)d_in[3];
  const float* conv2_w = (const float*)d_in[4];
  const float* conv2_b = (const float*)d_in[5];
  const float* enc_qw  = (const float*)d_in[6];
  const float* enc_qb  = (const float*)d_in[7];
  const float* enc_kw  = (const float*)d_in[8];
  const float* enc_kb  = (const float*)d_in[9];
  const float* enc_vw  = (const float*)d_in[10];
  const float* enc_vb  = (const float*)d_in[11];
  const float* enc_ow  = (const float*)d_in[12];
  const float* enc_ob  = (const float*)d_in[13];
  const float* enc_fcw = (const float*)d_in[14];
  const float* enc_fcb = (const float*)d_in[15];
  const float* enc_pjw = (const float*)d_in[16];
  const float* enc_pjb = (const float*)d_in[17];
  const float* sqw = (const float*)d_in[18];
  const float* sqb = (const float*)d_in[19];
  const float* skw = (const float*)d_in[20];
  const float* skb = (const float*)d_in[21];
  const float* svw = (const float*)d_in[22];
  const float* svb = (const float*)d_in[23];
  const float* sow = (const float*)d_in[24];
  const float* sob = (const float*)d_in[25];
  const float* cqw = (const float*)d_in[26];
  const float* cqb = (const float*)d_in[27];
  const float* ckw = (const float*)d_in[28];
  const float* ckb = (const float*)d_in[29];
  const float* cvw = (const float*)d_in[30];
  const float* cvb = (const float*)d_in[31];
  const float* cow = (const float*)d_in[32];
  const float* cob = (const float*)d_in[33];
  const float* fcw = (const float*)d_in[34];
  const float* fcb = (const float*)d_in[35];
  const float* pjw = (const float*)d_in[36];
  const float* pjb = (const float*)d_in[37];
  const float* emb = (const float*)d_in[38];
  const float* pe  = (const float*)d_in[39];
  float* out = (float*)d_out;

  char* p = (char*)d_ws;
  auto to_bf16 = [&](const float* src, size_t n) -> bf16* {
    bf16* dst = (bf16*)arena_alloc(p, n * sizeof(bf16));
    size_t blk = (n + 1023) / 1024;
    unsigned g = (unsigned)(blk < 8192 ? blk : 8192);
    f32_to_bf16_kernel<<<g, 256, 0, stream>>>(src, dst, (long long)n);
    return dst;
  };

  // --- weights -> bf16 ---
  bf16* bqw  = to_bf16(enc_qw, (size_t)2 * NE * NE);
  bf16* bkw  = to_bf16(enc_kw, (size_t)2 * NE * NE);
  bf16* bvw  = to_bf16(enc_vw, (size_t)2 * NE * NE);
  bf16* bow  = to_bf16(enc_ow, (size_t)2 * NE * NE);
  bf16* bfcw = to_bf16(enc_fcw, (size_t)2 * FF * NE);
  bf16* bpjw = to_bf16(enc_pjw, (size_t)2 * NE * FF);
  bf16* bsq  = to_bf16(sqw, (size_t)NE * NE);
  bf16* bsk  = to_bf16(skw, (size_t)NE * NE);
  bf16* bsv  = to_bf16(svw, (size_t)NE * NE);
  bf16* bso  = to_bf16(sow, (size_t)NE * NE);
  bf16* bcq  = to_bf16(cqw, (size_t)NE * NE);
  bf16* bck  = to_bf16(ckw, (size_t)NE * NE);
  bf16* bcv  = to_bf16(cvw, (size_t)NE * NE);
  bf16* bco  = to_bf16(cow, (size_t)NE * NE);
  bf16* bfc2 = to_bf16(fcw, (size_t)FF * NE);
  bf16* bpj2 = to_bf16(pjw, (size_t)NE * FF);
  bf16* bemb = to_bf16(emb, (size_t)NVOC * NE);

  // --- fp32 scratch ---
  float* x1  = (float*)arena_alloc(p, (size_t)BSZ * NE * TIN * 4);
  float* X   = (float*)arena_alloc(p, (size_t)M_ENC * NE * 4);
  float* tf  = (float*)arena_alloc(p, (size_t)M_ENC * NE * 4);
  float* qh  = (float*)arena_alloc(p, (size_t)M_ENC * NE * 4);
  float* kh  = (float*)arena_alloc(p, (size_t)M_ENC * NE * 4);
  float* vh  = (float*)arena_alloc(p, (size_t)M_ENC * NE * 4);
  float* ak  = (float*)arena_alloc(p, (size_t)M_ENC * NE * 4);
  float* av  = (float*)arena_alloc(p, (size_t)M_ENC * NE * 4);
  float* ctx = (float*)arena_alloc(p, (size_t)M_DEC * NE * 4);
  float* q0  = (float*)arena_alloc(p, (size_t)M_DEC * NE * 4);
  float* res = (float*)arena_alloc(p, (size_t)M_DEC * NE * 4);
  float* td  = (float*)arena_alloc(p, (size_t)M_DEC * NE * 4);
  float* dq  = (float*)arena_alloc(p, (size_t)M_DEC * NE * 4);
  float* dk  = (float*)arena_alloc(p, (size_t)M_DEC * NE * 4);
  float* dv  = (float*)arena_alloc(p, (size_t)M_DEC * NE * 4);
  float* q2  = (float*)arena_alloc(p, (size_t)M_DEC * NE * 4);
  // --- bf16 activation scratch ---
  bf16* hb   = (bf16*)arena_alloc(p, (size_t)M_ENC * NE * 2);
  bf16* ab   = (bf16*)arena_alloc(p, (size_t)M_ENC * NE * 2);
  bf16* fb   = (bf16*)arena_alloc(p, (size_t)M_ENC * FF * 2);
  bf16* audb = (bf16*)arena_alloc(p, (size_t)M_ENC * NE * 2);
  bf16* cxb  = (bf16*)arena_alloc(p, (size_t)M_DEC * NE * 2);
  bf16* q0b  = (bf16*)arena_alloc(p, (size_t)M_DEC * NE * 2);
  bf16* rb   = (bf16*)arena_alloc(p, (size_t)M_DEC * NE * 2);

  const float enc_scale = 0.125f;               // 1/sqrt(64)
  const float dec_scale = 0.17677669529663687f; // 1/sqrt(32)

  // ===================== Encoder =====================
  conv1_kernel<<<(unsigned)(((size_t)BSZ * NE * TIN) / 256), 256, 0, stream>>>(mels, conv1_w, conv1_b, x1);
  conv2_kernel<<<(unsigned)(((size_t)BSZ * NE * TENC) / 256), 256, 0, stream>>>(x1, conv2_w, conv2_b, tf);
  pos_kernel<<<(unsigned)(((size_t)BSZ * TENC * NE) / 256), 256, 0, stream>>>(tf, X);

  for (int i = 0; i < 2; ++i) {
    const bf16* wq = bqw + (size_t)i * NE * NE;
    const bf16* wk = bkw + (size_t)i * NE * NE;
    const bf16* wv = bvw + (size_t)i * NE * NE;
    const bf16* wo = bow + (size_t)i * NE * NE;
    const bf16* wf = bfcw + (size_t)i * FF * NE;
    const bf16* wp = bpjw + (size_t)i * NE * FF;

    rms512_kernel<<<M_ENC, 256, 0, stream>>>(X, hb, M_ENC);
    gemm(stream, hb, wq, enc_qb + i * NE, nullptr, tf, nullptr, M_ENC, NE, NE, 0);
    split_heads_kernel<<<dim3(TENC, EH, BSZ), 64, 0, stream>>>(tf, qh, TENC, EH, EHD, 1);
    gemm(stream, hb, wk, enc_kb + i * NE, nullptr, tf, nullptr, M_ENC, NE, NE, 0);
    split_heads_kernel<<<dim3(TENC, EH, BSZ), 64, 0, stream>>>(tf, kh, TENC, EH, EHD, 1);
    gemm(stream, hb, wv, enc_vb + i * NE, nullptr, tf, nullptr, M_ENC, NE, NE, 0);
    split_heads_kernel<<<dim3(TENC, EH, BSZ), 64, 0, stream>>>(tf, vh, TENC, EH, EHD, 0);
    attn_kernel<<<dim3(TENC, EH, BSZ), 256, 0, stream>>>(qh, kh, vh, nullptr, ab,
        EH, TENC, TENC, EHD, 0, 1, enc_scale);
    gemm(stream, ab, wo, enc_ob + i * NE, X, X, nullptr, M_ENC, NE, NE, 0);
    rms512_kernel<<<M_ENC, 256, 0, stream>>>(X, hb, M_ENC);
    gemm(stream, hb, wf, enc_fcb + i * FF, nullptr, nullptr, fb, M_ENC, FF, NE, 1 /*gelu*/);
    gemm(stream, fb, wp, enc_pjb + i * NE, X, X, nullptr, M_ENC, NE, FF, 0);
  }

  // ===================== Decoder =====================
  rms512_kernel<<<M_ENC, 256, 0, stream>>>(X, audb, M_ENC);
  gemm(stream, audb, bck, ckb, nullptr, tf, nullptr, M_ENC, NE, NE, 0);
  split_heads_kernel<<<dim3(TENC, DH, BSZ), 64, 0, stream>>>(tf, ak, TENC, DH, DHD, 1);
  gemm(stream, audb, bcv, cvb, nullptr, tf, nullptr, M_ENC, NE, NE, 0);
  split_heads_kernel<<<dim3(TENC, DH, BSZ), 64, 0, stream>>>(tf, av, TENC, DH, DHD, 0);

  ctx_kernel<<<(unsigned)(((size_t)BSZ * LDEC * NE) / 256), 256, 0, stream>>>(ids, emb, pe, ctx, q0);
  rms512_kernel<<<M_DEC, 256, 0, stream>>>(ctx, cxb, M_DEC);
  rms512_kernel<<<M_DEC, 256, 0, stream>>>(q0, q0b, M_DEC);

  gemm(stream, q0b, bsq, sqb, nullptr, td, nullptr, M_DEC, NE, NE, 0);
  rope_split_rms_kernel<<<dim3(LDEC, DH, BSZ), 32, 0, stream>>>(td, dq, LDEC, 0);
  gemm(stream, cxb, bsk, skb, nullptr, td, nullptr, M_DEC, NE, NE, 0);
  rope_split_rms_kernel<<<dim3(LDEC, DH, BSZ), 32, 0, stream>>>(td, dk, LDEC, 1);
  gemm(stream, cxb, bsv, svb, nullptr, td, nullptr, M_DEC, NE, NE, 0);
  split_heads_kernel<<<dim3(LDEC, DH, BSZ), 64, 0, stream>>>(td, dv, LDEC, DH, DHD, 0);

  attn_kernel<<<dim3(LDEC, DH, BSZ), 256, 0, stream>>>(dq, dk, dv, ids, ab,
      DH, LDEC, LDEC, DHD, 1, 1, dec_scale);
  gemm(stream, ab, bso, sob, q0, res, nullptr, M_DEC, NE, NE, 0);

  rms512_kernel<<<M_DEC, 256, 0, stream>>>(res, rb, M_DEC);
  gemm(stream, rb, bcq, cqb, nullptr, td, nullptr, M_DEC, NE, NE, 0);
  rope_split_rms_kernel<<<dim3(LDEC, DH, BSZ), 32, 0, stream>>>(td, q2, LDEC, 0);
  attn_kernel<<<dim3(LDEC, DH, BSZ), 256, 0, stream>>>(q2, ak, av, nullptr, ab,
      DH, LDEC, TENC, DHD, 0, 0, dec_scale);
  gemm(stream, ab, bco, cob, res, res, nullptr, M_DEC, NE, NE, 0);

  rms512_kernel<<<M_DEC, 256, 0, stream>>>(res, rb, M_DEC);
  gemm(stream, rb, bfc2, fcb, nullptr, nullptr, fb, M_DEC, FF, NE, 1 /*gelu*/);
  gemm(stream, fb, bpj2, pjb, res, res, nullptr, M_DEC, NE, FF, 0);

  rms512_kernel<<<M_DEC, 256, 0, stream>>>(res, rb, M_DEC);
  gemm(stream, rb, bemb, nullptr, nullptr, out, nullptr, M_DEC, NVOC, NE, 0);
}
